// MoSARA_74826920231188
// MI455X (gfx1250) — matrix operand order repaired and checked by reading, compiled, not run on hardware
//
#include <hip/hip_runtime.h>
#include <math.h>

// Problem constants (match reference)
#define Bq 4096
#define Dq 2048
#define Kq 512
#define Eq 8

typedef __attribute__((ext_vector_type(16))) __bf16 v16bf;
typedef __attribute__((ext_vector_type(8)))  float  v8f;
typedef __attribute__((ext_vector_type(4)))  int    v4i_t;

#define AS1 __attribute__((address_space(1)))
#define AS3 __attribute__((address_space(3)))

#if defined(__HIP_DEVICE_COMPILE__) && __has_builtin(__builtin_amdgcn_global_load_async_to_lds_b128)
#define USE_ASYNC_LDS 1
#else
#define USE_ASYNC_LDS 0
#endif

__device__ __forceinline__ unsigned short f32_bf16(float f) {
    unsigned int u = __float_as_uint(f);
    u += 0x7FFFu + ((u >> 16) & 1u);   // round to nearest even
    return (unsigned short)(u >> 16);
}

#if USE_ASYNC_LDS
__device__ __forceinline__ void wait_async0() {
#if __has_builtin(__builtin_amdgcn_s_wait_asynccnt)
    __builtin_amdgcn_s_wait_asynccnt(0);
#else
    asm volatile("s_wait_asynccnt 0x0" ::: "memory");
#endif
}
#endif

// ---------------- elementwise f32 -> bf16 (4 per thread) ----------------
__global__ void cvt_bf16_kernel(const float* __restrict__ in,
                                unsigned short* __restrict__ out, int n4) {
    int i = blockIdx.x * blockDim.x + threadIdx.x;
    if (i >= n4) return;
    const float4 f = reinterpret_cast<const float4*>(in)[i];
    ushort4 o;
    o.x = f32_bf16(f.x); o.y = f32_bf16(f.y);
    o.z = f32_bf16(f.z); o.w = f32_bf16(f.w);
    reinterpret_cast<ushort4*>(out)[i] = o;
}

// ---------------- transposed f32 (R x C) -> bf16 (C x R) ----------------
__global__ void tcvt_bf16_kernel(const float* __restrict__ in,
                                 unsigned short* __restrict__ out, int R, int C) {
    __shared__ float tile[32][33];
    int c = blockIdx.x * 32 + threadIdx.x;
    int r0 = blockIdx.y * 32;
#pragma unroll
    for (int j = 0; j < 32; j += 8) {
        tile[threadIdx.y + j][threadIdx.x] = in[(size_t)(r0 + threadIdx.y + j) * C + c];
    }
    __syncthreads();
    int orow0 = blockIdx.x * 32;
#pragma unroll
    for (int j = 0; j < 32; j += 8) {
        out[(size_t)(orow0 + threadIdx.y + j) * R + r0 + threadIdx.x] =
            f32_bf16(tile[threadIdx.x][threadIdx.y + j]);
    }
}

// ---------------- bf16 WMMA GEMM: C[M,N] = A[M,Kd] * Bm[Kd,N] ----------------
// MODE 0: C = acc
// MODE 1: C = (C + acc) * (1 + vscale[n])
#define BMT 128
#define BNT 128
#define BKT 32
#define LDP 40   // padded LDS row stride (bf16 elems), keeps 16B alignment

struct BT { uint4 lo, hi; };     // two consecutive k-rows, 8 cols each

__device__ __forceinline__ BT loadB(const unsigned short* __restrict__ Bm,
                                    int N, int n0, int kk, int k2, int c8) {
    BT t;
    t.lo = *reinterpret_cast<const uint4*>(Bm + (size_t)(kk + k2) * N + n0 + c8);
    t.hi = *reinterpret_cast<const uint4*>(Bm + (size_t)(kk + k2 + 1) * N + n0 + c8);
    return t;
}

// pack (k2,k2+1) bf16 pairs into dwords -> Bs[n][k] transposed layout, b32 stores
__device__ __forceinline__ void storeB(unsigned short* Bsbuf, int k2, int c8, BT t) {
    const unsigned short* e0 = reinterpret_cast<const unsigned short*>(&t.lo);
    const unsigned short* e1 = reinterpret_cast<const unsigned short*>(&t.hi);
#pragma unroll
    for (int i = 0; i < 8; ++i) {
        unsigned d = (unsigned)e0[i] | ((unsigned)e1[i] << 16);
        *reinterpret_cast<unsigned*>(Bsbuf + (c8 + i) * LDP + k2) = d;
    }
}

#if USE_ASYNC_LDS
__device__ __forceinline__ void issueA(const unsigned short* __restrict__ A,
                                       int Kd, int m0, int kk, int row, int col8,
                                       unsigned short* Asbuf) {
#pragma unroll
    for (int it = 0; it < 2; ++it) {
        int r = row + it * 64;
        const unsigned short* gp = A + (size_t)(m0 + r) * Kd + kk + col8;
        unsigned short* lp = Asbuf + r * LDP + col8;
        __builtin_amdgcn_global_load_async_to_lds_b128(
            (AS1 v4i_t*)gp, (AS3 v4i_t*)lp, 0, 0);
    }
}
#else
struct AT { uint4 a0, a1; };
__device__ __forceinline__ AT loadA(const unsigned short* __restrict__ A,
                                    int Kd, int m0, int kk, int row, int col8) {
    AT t;
    t.a0 = *reinterpret_cast<const uint4*>(A + (size_t)(m0 + row) * Kd + kk + col8);
    t.a1 = *reinterpret_cast<const uint4*>(A + (size_t)(m0 + row + 64) * Kd + kk + col8);
    return t;
}
__device__ __forceinline__ void storeA(unsigned short* Asbuf, int row, int col8, AT t) {
    *reinterpret_cast<uint4*>(Asbuf + row * LDP + col8)        = t.a0;
    *reinterpret_cast<uint4*>(Asbuf + (row + 64) * LDP + col8) = t.a1;
}
#endif

template <int MODE>
__global__ __launch_bounds__(256)
void wmma_gemm_kernel(const unsigned short* __restrict__ A,
                      const unsigned short* __restrict__ Bm,
                      float* __restrict__ C,
                      int M, int N, int Kd,
                      const float* __restrict__ vscale) {
    __shared__ unsigned short As[2][BMT * LDP];   // [m][k]
    __shared__ unsigned short Bs[2][BNT * LDP];   // [n][k] (transposed on store)

    const int tid  = threadIdx.x;
    const int lane = tid & 31;
    const int wave = tid >> 5;        // 8 waves
    const int wm   = wave >> 1;       // 0..3 -> 32-row strip
    const int wn   = wave & 1;        // 0..1 -> 64-col strip
    const int m0   = blockIdx.y * BMT;
    const int n0   = blockIdx.x * BNT;

    v8f acc[2][4];
#pragma unroll
    for (int i = 0; i < 2; ++i)
#pragma unroll
        for (int j = 0; j < 4; ++j) {
            v8f z = {0.f, 0.f, 0.f, 0.f, 0.f, 0.f, 0.f, 0.f};
            acc[i][j] = z;
        }

    // per-lane fragment coordinates (ISA 7.12.2 layouts, wave32)
    const int l15 = lane & 15;
    const int hi  = lane >> 4;
    const int akb = hi * 8;     // A: K halves 0..7 / 8..15 (+16 for 2nd group)
    const int bkb = hi * 8;     // B: K halves 0..7 / 8..15 (+16 for 2nd group)
    int arow[2];
    arow[0] = wm * 32 + l15;
    arow[1] = arow[0] + 16;

    // staging coordinates
    const int sa_row  = tid >> 2;        // A: 0..63 (x2 via +64)
    const int sa_col8 = (tid & 3) * 8;
    const int sb_k2   = (tid >> 4) * 2;  // B: even k 0..30
    const int sb_c8   = (tid & 15) * 8;  // B: cols 0..120

    // ---- prologue: stage tile 0 into buffer 0 ----
    {
        BT bt = loadB(Bm, N, n0, 0, sb_k2, sb_c8);
#if USE_ASYNC_LDS
        issueA(A, Kd, m0, 0, sa_row, sa_col8, As[0]);
#else
        AT at = loadA(A, Kd, m0, 0, sa_row, sa_col8);
#endif
        storeB(Bs[0], sb_k2, sb_c8, bt);
#if USE_ASYNC_LDS
        wait_async0();
#else
        storeA(As[0], sa_row, sa_col8, at);
#endif
    }
    __syncthreads();

    int buf = 0;
    for (int kk = 0; kk < Kd; kk += BKT) {
        const int  nbuf = buf ^ 1;
        const bool more = (kk + BKT) < Kd;

        // prefetch next tile (global -> regs / async -> LDS) before compute
        BT nbt;
#if !USE_ASYNC_LDS
        AT nat;
#endif
        if (more) {
            nbt = loadB(Bm, N, n0, kk + BKT, sb_k2, sb_c8);
#if USE_ASYNC_LDS
            issueA(A, Kd, m0, kk + BKT, sa_row, sa_col8, As[nbuf]);
#else
            nat = loadA(A, Kd, m0, kk + BKT, sa_row, sa_col8);
#endif
        }

        // compute on current buffer
        union Frag { v16bf v; uint4 q[2]; };
        Frag afr[2];
#pragma unroll
        for (int i = 0; i < 2; ++i) {
            const unsigned short* p = &As[buf][arow[i] * LDP + akb];
            afr[i].q[0] = *reinterpret_cast<const uint4*>(p);        // K = kb..kb+7
            afr[i].q[1] = *reinterpret_cast<const uint4*>(p + 16);   // K = 16+kb..
        }
#pragma unroll
        for (int j = 0; j < 4; ++j) {
            const int bcol = wn * 64 + j * 16 + l15;
            const unsigned short* p = &Bs[buf][bcol * LDP + bkb];
            Frag bfr;
            bfr.q[0] = *reinterpret_cast<const uint4*>(p);
            bfr.q[1] = *reinterpret_cast<const uint4*>(p + 16);
#pragma unroll
            for (int i = 0; i < 2; ++i) {
                acc[i][j] = __builtin_amdgcn_wmma_f32_16x16x32_bf16(
                    false, afr[i].v, false, bfr.v, (short)0, acc[i][j],
                    false, false);
            }
        }

        // commit next tile to LDS
        if (more) {
            storeB(Bs[nbuf], sb_k2, sb_c8, nbt);
#if USE_ASYNC_LDS
            wait_async0();
#else
            storeA(As[nbuf], sa_row, sa_col8, nat);
#endif
        }
        __syncthreads();
        buf = nbuf;
    }

    // epilogue: C/D layout -> col = lane&15, row = q + (lane>>4)*8
    const int crow = hi * 8;
#pragma unroll
    for (int j = 0; j < 4; ++j) {
        const int col = n0 + wn * 64 + j * 16 + l15;
        float vs = 1.0f;
        if (MODE == 1) vs = 1.0f + vscale[col];
#pragma unroll
        for (int i = 0; i < 2; ++i) {
#pragma unroll
            for (int q = 0; q < 8; ++q) {
                int rowg = m0 + wm * 32 + i * 16 + crow + q;
                size_t idx = (size_t)rowg * N + col;
                float val = acc[i][j][q];
                if (MODE == 1) val = (C[idx] + val) * vs;
                C[idx] = val;
            }
        }
    }
}

// ------------- router + gating: y[b,k] = t[b,k] * sum_e softmax(s_b*W2)[e]*lam[e,k] -------------
__global__ __launch_bounds__(256)
void router_kernel(const float* __restrict__ r, const float* __restrict__ t,
                   const float* __restrict__ W1, const float* __restrict__ W2,
                   const float* __restrict__ lam, unsigned short* __restrict__ y) {
    const int b = blockIdx.x;
    const int tid = threadIdx.x;
    __shared__ float red[256];
    __shared__ float g[Eq];

    float p = 0.f;
    for (int k = tid; k < Kq; k += 256) p += r[(size_t)b * Kq + k] * W1[k];
    red[tid] = p;
    __syncthreads();
    for (int off = 128; off > 0; off >>= 1) {
        if (tid < off) red[tid] += red[tid + off];
        __syncthreads();
    }
    if (tid == 0) {
        float s = red[0];
        float l[Eq], mx = -3.4e38f;
#pragma unroll
        for (int e = 0; e < Eq; ++e) { l[e] = s * W2[e]; mx = fmaxf(mx, l[e]); }
        float sum = 0.f;
#pragma unroll
        for (int e = 0; e < Eq; ++e) { l[e] = __expf(l[e] - mx); sum += l[e]; }
        float inv = 1.0f / sum;
#pragma unroll
        for (int e = 0; e < Eq; ++e) g[e] = l[e] * inv;
    }
    __syncthreads();

    for (int k = tid; k < Kq; k += 256) {
        float lamk = 0.f;
#pragma unroll
        for (int e = 0; e < Eq; ++e) lamk += g[e] * lam[(size_t)e * Kq + k];
        y[(size_t)b * Kq + k] = f32_bf16(t[(size_t)b * Kq + k] * lamk);
    }
}

// ---------------------------------------------------------------------------
extern "C" void kernel_launch(void* const* d_in, const int* in_sizes, int n_in,
                              void* d_out, int out_size, void* d_ws, size_t ws_size,
                              hipStream_t stream) {
    (void)in_sizes; (void)n_in; (void)out_size; (void)ws_size;
    const float* x   = (const float*)d_in[0];   // (B,D)
    const float* W   = (const float*)d_in[1];   // (D,D)
    const float* U   = (const float*)d_in[2];   // (D,K)
    const float* V   = (const float*)d_in[3];   // (K,D)
    const float* lam = (const float*)d_in[4];   // (E,K)
    const float* v   = (const float*)d_in[5];   // (D,)
    const float* W1  = (const float*)d_in[6];   // (K,1)
    const float* W2  = (const float*)d_in[7];   // (1,E)
    float* out = (float*)d_out;                 // (B,D)

    char* ws = (char*)d_ws;
    size_t off = 0;
    auto alloc = [&](size_t bytes) -> void* {
        void* p = ws + off;
        off += (bytes + 255) & ~(size_t)255;
        return p;
    };
    unsigned short* xh = (unsigned short*)alloc((size_t)Bq * Dq * 2); // bf16 x
    unsigned short* Wt = (unsigned short*)alloc((size_t)Dq * Dq * 2); // bf16 W^T  [d][n]
    unsigned short* Uh = (unsigned short*)alloc((size_t)Dq * Kq * 2); // bf16 U    [d][k]
    unsigned short* Ut = (unsigned short*)alloc((size_t)Kq * Dq * 2); // bf16 U^T  [k][n]
    unsigned short* Vt = (unsigned short*)alloc((size_t)Dq * Kq * 2); // bf16 V^T  [d][k]
    float*          rr = (float*)alloc((size_t)Bq * Kq * 4);          // routing = x@U
    float*          tt = (float*)alloc((size_t)Bq * Kq * 4);          // t = x@V^T
    unsigned short* yy = (unsigned short*)alloc((size_t)Bq * Kq * 2); // bf16 gated t

    // 1) precision conversions / transposes
    {
        int n4 = (Bq * Dq) / 4;
        cvt_bf16_kernel<<<n4 / 256, 256, 0, stream>>>(x, xh, n4);
    }
    {
        int n4 = (Dq * Kq) / 4;
        cvt_bf16_kernel<<<n4 / 256, 256, 0, stream>>>(U, Uh, n4);
    }
    {
        dim3 blk(32, 8);
        tcvt_bf16_kernel<<<dim3(Dq / 32, Dq / 32), blk, 0, stream>>>(W, Wt, Dq, Dq);
        tcvt_bf16_kernel<<<dim3(Kq / 32, Dq / 32), blk, 0, stream>>>(U, Ut, Dq, Kq);
        tcvt_bf16_kernel<<<dim3(Dq / 32, Kq / 32), blk, 0, stream>>>(V, Vt, Kq, Dq);
    }

    // 2) base = x @ W^T  -> d_out      (M=B, N=D, Kd=D)
    wmma_gemm_kernel<0><<<dim3(Dq / BNT, Bq / BMT), 256, 0, stream>>>(
        xh, Wt, out, Bq, Dq, Dq, nullptr);

    // 3) r = x @ U  ;  t = x @ V^T     (M=B, N=K, Kd=D)
    wmma_gemm_kernel<0><<<dim3(Kq / BNT, Bq / BMT), 256, 0, stream>>>(
        xh, Uh, rr, Bq, Kq, Dq, nullptr);
    wmma_gemm_kernel<0><<<dim3(Kq / BNT, Bq / BMT), 256, 0, stream>>>(
        xh, Vt, tt, Bq, Kq, Dq, nullptr);

    // 4) router softmax + gating -> y (bf16)
    router_kernel<<<Bq, 256, 0, stream>>>(rr, tt, W1, W2, lam, yy);

    // 5) out = (base + y @ U^T) * (1+v)   (M=B, N=D, Kd=K)
    wmma_gemm_kernel<1><<<dim3(Dq / BNT, Bq / BMT), 256, 0, stream>>>(
        yy, Ut, out, Bq, Dq, Kq, v);
}
